// MoEMLP_37374805410167
// MI455X (gfx1250) — matrix operand order repaired and checked by reading, not compiled, not run on hardware
//
#include <hip/hip_runtime.h>
#include <hip/hip_bf16.h>
#include <math.h>

// MoE top-2 of 8 experts, N=4096 tokens, d=1024, hidden=4096.
// Sparse routing (only top-2 experts computed) + f16 WMMA GEMMs w/ f32 accum.
// A tiles move via CDNA5 async global->LDS b128 copies (ASYNCcnt), B tiles are
// fp32->f16 converted + transposed into LDS on the fly (co-executes with WMMA).
// All loader addresses are loop-invariant running pointers (+= const stride).
#define NTOK   4096
#define DIM    1024
#define HID    4096
#define NE     8
#define TOPK   2
#define BM     128
#define BN     128
#define BK     32
#define ROWTILES 72            // worst-case padded row tiles: 8192 + 8*127 -> 9216/128
#define MAXROWS (ROWTILES*BM)  // 9216
#define LDST   40              // padded LDS stride (halves): conflict-free b128 frag loads

typedef _Float16 v16h __attribute__((ext_vector_type(16)));
typedef _Float16 v8h  __attribute__((ext_vector_type(8)));
typedef float    v8f  __attribute__((ext_vector_type(8)));
typedef int      i32x4 __attribute__((ext_vector_type(4)));

union Frag { v16h v; v8h h[2]; };

#if defined(__HIP_DEVICE_COMPILE__) && __has_builtin(__builtin_amdgcn_global_load_async_to_lds_b128)
#define HAVE_ASYNC_LDS 1
#endif

// 16-byte global -> LDS copy. Async path uses GLOBAL_LOAD_ASYNC_TO_LDS_B128.
__device__ __forceinline__ void cp16(void* lds, const void* g) {
#ifdef HAVE_ASYNC_LDS
  __builtin_amdgcn_global_load_async_to_lds_b128(
      (__attribute__((address_space(1))) i32x4*)g,
      (__attribute__((address_space(3))) i32x4*)lds, 0, 0);
#else
  *(v8h*)lds = *(const v8h*)g;
#endif
}
__device__ __forceinline__ void wait_async() {
#ifdef HAVE_ASYNC_LDS
#if __has_builtin(__builtin_amdgcn_s_wait_asynccnt)
  __builtin_amdgcn_s_wait_asynccnt(0);
#else
  asm volatile("s_wait_asynccnt 0x0" ::: "memory");
#endif
#endif
}

// A fragment (16x32 f16, M x K): lanes 0-15 -> K 0..7 & 16..23, lanes 16-31 -> K 8..15 & 24..31
__device__ __forceinline__ v16h frag_a(const _Float16* As, int lane, int mrow) {
  int l15 = lane & 15, hi = lane >> 4;
  const _Float16* p = As + (mrow + l15) * LDST + hi * 8;
  Frag u; u.h[0] = *(const v8h*)p; u.h[1] = *(const v8h*)(p + 16);
  return u.v;
}
// B fragment (32x16 f16, K x N) from K-contiguous (transposed) LDS tile:
// lanes 0-15 -> K 0..7 & 16..23, lanes 16-31 -> K 8..15 & 24..31, N = lane&15
__device__ __forceinline__ v16h frag_b(const _Float16* Bt, int lane, int ncol) {
  int l15 = lane & 15, hi = lane >> 4;
  const _Float16* p = Bt + (ncol + l15) * LDST + hi * 8;
  Frag u; u.h[0] = *(const v8h*)p; u.h[1] = *(const v8h*)(p + 16);
  return u.v;
}

__device__ __forceinline__ float gelu_exact(float v) {
  return 0.5f * v * (1.0f + erff(v * 0.70710678118654752f));
}

// 8 independent WMMAs for one 32x64 wave tile (2 A frags x 4 B frags)
__device__ __forceinline__ void wmma8(v8f acc[2][4], const v16h& a0, const v16h& a1,
                                      const v16h& b0, const v16h& b1,
                                      const v16h& b2, const v16h& b3) {
  acc[0][0] = __builtin_amdgcn_wmma_f32_16x16x32_f16(false, a0, false, b0, (short)0, acc[0][0], false, false);
  acc[1][0] = __builtin_amdgcn_wmma_f32_16x16x32_f16(false, a1, false, b0, (short)0, acc[1][0], false, false);
  acc[0][1] = __builtin_amdgcn_wmma_f32_16x16x32_f16(false, a0, false, b1, (short)0, acc[0][1], false, false);
  acc[1][1] = __builtin_amdgcn_wmma_f32_16x16x32_f16(false, a1, false, b1, (short)0, acc[1][1], false, false);
  acc[0][2] = __builtin_amdgcn_wmma_f32_16x16x32_f16(false, a0, false, b2, (short)0, acc[0][2], false, false);
  acc[1][2] = __builtin_amdgcn_wmma_f32_16x16x32_f16(false, a1, false, b2, (short)0, acc[1][2], false, false);
  acc[0][3] = __builtin_amdgcn_wmma_f32_16x16x32_f16(false, a0, false, b3, (short)0, acc[0][3], false, false);
  acc[1][3] = __builtin_amdgcn_wmma_f32_16x16x32_f16(false, a1, false, b3, (short)0, acc[1][3], false, false);
}

// ---------------- x -> f16 copy (makes FFN1 A tiles pure async copies) ----------------
__global__ void k_xcvt(const float* __restrict__ x, _Float16* __restrict__ xh) {
  size_t i = ((size_t)blockIdx.x * 256 + threadIdx.x) * 8;
  float4 a = *(const float4*)(x + i);
  float4 b = *(const float4*)(x + i + 4);
  v8h o;
  o[0] = (_Float16)a.x; o[1] = (_Float16)a.y; o[2] = (_Float16)a.z; o[3] = (_Float16)a.w;
  o[4] = (_Float16)b.x; o[5] = (_Float16)b.y; o[6] = (_Float16)b.z; o[7] = (_Float16)b.w;
  *(v8h*)(xh + i) = o;
}

// ---------------- Router: logits, softmax, top-2, counts ----------------
__global__ void k_router(const float* __restrict__ x, const float* __restrict__ wg,
                         float* __restrict__ probs, int* __restrict__ topi,
                         float* __restrict__ topw, int* __restrict__ counts) {
  __shared__ float red[128 * NE];
  const int n = blockIdx.x, tid = threadIdx.x;
  float part[NE] = {0};
  const float* xr = x + (size_t)n * DIM;
  for (int c = tid; c < DIM; c += 128) {
    float xv = xr[c];
    const float* wr = wg + c * NE;
    #pragma unroll
    for (int e = 0; e < NE; ++e) part[e] += xv * wr[e];
  }
  #pragma unroll
  for (int e = 0; e < NE; ++e) red[tid * NE + e] = part[e];
  __syncthreads();
  for (int s = 64; s > 0; s >>= 1) {
    if (tid < s) {
      #pragma unroll
      for (int e = 0; e < NE; ++e) red[tid * NE + e] += red[(tid + s) * NE + e];
    }
    __syncthreads();
  }
  if (tid == 0) {
    float l[NE], p[NE];
    float mx = red[0];
    #pragma unroll
    for (int e = 0; e < NE; ++e) { l[e] = red[e]; mx = fmaxf(mx, l[e]); }
    float s = 0.f;
    #pragma unroll
    for (int e = 0; e < NE; ++e) { p[e] = expf(l[e] - mx); s += p[e]; }
    float inv = 1.f / s;
    #pragma unroll
    for (int e = 0; e < NE; ++e) { p[e] *= inv; probs[(size_t)n * NE + e] = p[e]; }
    int i1 = 0;
    #pragma unroll
    for (int e = 1; e < NE; ++e) if (p[e] > p[i1]) i1 = e;
    int i2 = (i1 == 0) ? 1 : 0;
    #pragma unroll
    for (int e = 0; e < NE; ++e) if (e != i1 && p[e] > p[i2]) i2 = e;
    float sw = p[i1] + p[i2];
    topi[n * TOPK + 0] = i1; topi[n * TOPK + 1] = i2;
    topw[n * TOPK + 0] = p[i1] / sw; topw[n * TOPK + 1] = p[i2] / sw;
    atomicAdd(&counts[i1], 1);
    atomicAdd(&counts[i2], 1);
  }
}

// ---------------- Scan: tile-aligned per-expert offsets ----------------
__global__ void k_scan(const int* __restrict__ counts, int* __restrict__ offs,
                       int* __restrict__ cursors) {
  if (threadIdx.x == 0) {
    int acc = 0;
    for (int e = 0; e < NE; ++e) {
      offs[e] = acc;
      cursors[e] = acc;
      acc += (counts[e] + BM - 1) & ~(BM - 1);
    }
    offs[NE] = acc;   // padded total row count
  }
}

// ---------------- Scatter tokens into per-expert lists ----------------
__global__ void k_scatter(const int* __restrict__ topi, const float* __restrict__ topw,
                          int* __restrict__ cursors, int* __restrict__ tokl,
                          float* __restrict__ tokw) {
  int n = blockIdx.x * blockDim.x + threadIdx.x;
  if (n >= NTOK) return;
  #pragma unroll
  for (int k = 0; k < TOPK; ++k) {
    int e = topi[n * TOPK + k];
    int pos = atomicAdd(&cursors[e], 1);
    tokl[pos] = n;
    tokw[pos] = topw[n * TOPK + k];
  }
}

// ---------------- FFN1: H = gelu(gather(Xh) @ W1_e), f16 out ----------------
__launch_bounds__(256)
__global__ void k_ffn1(const _Float16* __restrict__ xh, const float* __restrict__ w1,
                       const int* __restrict__ tokl, const int* __restrict__ offs,
                       const _Float16* __restrict__ zbuf, _Float16* __restrict__ H) {
  __shared__ alignas(16) _Float16 As[2][BM * LDST];
  __shared__ alignas(16) _Float16 Bt[2][BN * LDST];
  __shared__ int s_tok[BM];

  const int tid = threadIdx.x;
  const int row0 = blockIdx.x * BM;
  const int Mtot = offs[NE];
  if (row0 >= Mtot) return;
  const int n0 = blockIdx.y * BN;

  int e = NE - 1;
  #pragma unroll
  for (int i = NE - 1; i >= 0; --i) if (row0 < offs[i + 1]) e = i;

  if (tid < BM) s_tok[tid] = tokl[row0 + tid];
  __syncthreads();

  // ---- loop-invariant loader state (running pointers) ----
  // A: 2 async b128 copies per thread (128 rows x 4 chunks / 256 threads)
  const int rA0 = tid >> 2,          cA0 = tid & 3;
  const int rA1 = (tid + 256) >> 2,  cA1 = (tid + 256) & 3;
  const int tA0 = s_tok[rA0],        tA1 = s_tok[rA1];
  const _Float16* gA0 = (tA0 >= 0) ? xh + (size_t)tA0 * DIM + cA0 * 8 : zbuf;
  const _Float16* gA1 = (tA1 >= 0) ? xh + (size_t)tA1 * DIM + cA1 * 8 : zbuf;
  const int stA0 = (tA0 >= 0) ? BK : 0;
  const int stA1 = (tA1 >= 0) ? BK : 0;
  const int dA0 = rA0 * LDST + cA0 * 8;
  const int dA1 = rA1 * LDST + cA1 * 8;
  // B: 4 float4 loads per thread (32 k-rows x 32 float4 / 256 threads)
  const float* w1e = w1 + (size_t)e * DIM * HID + n0;
  const float* gB[4];
  int dB[4];
  #pragma unroll
  for (int it = 0; it < 4; ++it) {
    int idx = tid + it * 256;
    int k = idx >> 5, n4 = idx & 31;
    gB[it] = w1e + (size_t)k * HID + n4 * 4;
    dB[it] = (n4 * 4) * LDST + k;
  }

  auto loadA = [&](int buf) {
    cp16(&As[buf][dA0], gA0); gA0 += stA0;
    cp16(&As[buf][dA1], gA1); gA1 += stA1;
  };
  auto loadB = [&](int buf) {
    #pragma unroll
    for (int it = 0; it < 4; ++it) {
      __builtin_prefetch(gB[it] + (size_t)2 * BK * HID, 0, 1);  // speculative: safe off-end
      float4 v = *(const float4*)gB[it];
      gB[it] += (size_t)BK * HID;
      _Float16* d = &Bt[buf][dB[it]];   // transposed: [n][k]
      d[0 * LDST] = (_Float16)v.x; d[1 * LDST] = (_Float16)v.y;
      d[2 * LDST] = (_Float16)v.z; d[3 * LDST] = (_Float16)v.w;
    }
  };

  const int lane = tid & 31, wid = tid >> 5;
  const int wm = (wid & 3) * 32, wn = (wid >> 2) * 64;   // wave: 32 rows x 64 cols
  v8f acc[2][4] = {};

  loadA(0); loadB(0);
  wait_async();
  __syncthreads();
  const int NK = DIM / BK;
  for (int kk = 0; kk < NK; ++kk) {
    int cur = kk & 1;
    if (kk + 1 < NK) { loadA(cur ^ 1); loadB(cur ^ 1); }
    v16h a0 = frag_a(As[cur], lane, wm);
    v16h a1 = frag_a(As[cur], lane, wm + 16);
    v16h b0 = frag_b(Bt[cur], lane, wn);
    v16h b1 = frag_b(Bt[cur], lane, wn + 16);
    v16h b2 = frag_b(Bt[cur], lane, wn + 32);
    v16h b3 = frag_b(Bt[cur], lane, wn + 48);
    wmma8(acc, a0, a1, b0, b1, b2, b3);
    wait_async();
    __syncthreads();
  }

  const int l15 = lane & 15, hi = lane >> 4;
  #pragma unroll
  for (int m = 0; m < 2; ++m)
    #pragma unroll
    for (int n = 0; n < 4; ++n) {
      int col = n0 + wn + n * 16 + l15;
      size_t base = (size_t)(row0 + wm + m * 16 + hi * 8) * HID + col;
      #pragma unroll
      for (int r = 0; r < 8; ++r)
        H[base + (size_t)r * HID] = (_Float16)gelu_exact(acc[m][n][r]);
    }
}

// ---------------- FFN2: out[token] += w * (H @ W2_e) ----------------
__launch_bounds__(256)
__global__ void k_ffn2(const _Float16* __restrict__ H, const float* __restrict__ w2,
                       const int* __restrict__ tokl, const float* __restrict__ tokw,
                       const int* __restrict__ offs, float* __restrict__ out) {
  __shared__ alignas(16) _Float16 As[2][BM * LDST];
  __shared__ alignas(16) _Float16 Bt[2][BN * LDST];
  __shared__ int s_tok[BM];
  __shared__ float s_w[BM];

  const int tid = threadIdx.x;
  const int row0 = blockIdx.x * BM;
  const int Mtot = offs[NE];
  if (row0 >= Mtot) return;
  const int n0 = blockIdx.y * BN;

  int e = NE - 1;
  #pragma unroll
  for (int i = NE - 1; i >= 0; --i) if (row0 < offs[i + 1]) e = i;

  if (tid < BM) { s_tok[tid] = tokl[row0 + tid]; s_w[tid] = tokw[row0 + tid]; }

  // ---- loop-invariant loader state ----
  const int rA0 = tid >> 2,          cA0 = tid & 3;
  const int rA1 = (tid + 256) >> 2,  cA1 = (tid + 256) & 3;
  const _Float16* gA0 = H + (size_t)(row0 + rA0) * HID + cA0 * 8;
  const _Float16* gA1 = H + (size_t)(row0 + rA1) * HID + cA1 * 8;
  const int dA0 = rA0 * LDST + cA0 * 8;
  const int dA1 = rA1 * LDST + cA1 * 8;
  const float* w2e = w2 + (size_t)e * HID * DIM + n0;
  const float* gB[4];
  int dB[4];
  #pragma unroll
  for (int it = 0; it < 4; ++it) {
    int idx = tid + it * 256;
    int k = idx >> 5, n4 = idx & 31;
    gB[it] = w2e + (size_t)k * DIM + n4 * 4;
    dB[it] = (n4 * 4) * LDST + k;
  }

  auto loadA = [&](int buf) {
    cp16(&As[buf][dA0], gA0); gA0 += BK;
    cp16(&As[buf][dA1], gA1); gA1 += BK;
  };
  auto loadB = [&](int buf) {
    #pragma unroll
    for (int it = 0; it < 4; ++it) {
      __builtin_prefetch(gB[it] + (size_t)2 * BK * DIM, 0, 1);
      float4 v = *(const float4*)gB[it];
      gB[it] += (size_t)BK * DIM;
      _Float16* d = &Bt[buf][dB[it]];   // transposed: [n][k]
      d[0 * LDST] = (_Float16)v.x; d[1 * LDST] = (_Float16)v.y;
      d[2 * LDST] = (_Float16)v.z; d[3 * LDST] = (_Float16)v.w;
    }
  };

  const int lane = tid & 31, wid = tid >> 5;
  const int wm = (wid & 3) * 32, wn = (wid >> 2) * 64;
  v8f acc[2][4] = {};

  loadA(0); loadB(0);
  wait_async();
  __syncthreads();
  const int NK = HID / BK;
  for (int kk = 0; kk < NK; ++kk) {
    int cur = kk & 1;
    if (kk + 1 < NK) { loadA(cur ^ 1); loadB(cur ^ 1); }
    v16h a0 = frag_a(As[cur], lane, wm);
    v16h a1 = frag_a(As[cur], lane, wm + 16);
    v16h b0 = frag_b(Bt[cur], lane, wn);
    v16h b1 = frag_b(Bt[cur], lane, wn + 16);
    v16h b2 = frag_b(Bt[cur], lane, wn + 32);
    v16h b3 = frag_b(Bt[cur], lane, wn + 48);
    wmma8(acc, a0, a1, b0, b1, b2, b3);
    wait_async();
    __syncthreads();
  }

  const int l15 = lane & 15, hi = lane >> 4;
  #pragma unroll
  for (int m = 0; m < 2; ++m)
    #pragma unroll
    for (int n = 0; n < 4; ++n) {
      int col = n0 + wn + n * 16 + l15;
      #pragma unroll
      for (int r = 0; r < 8; ++r) {
        int rl = wm + m * 16 + hi * 8 + r;
        int t = s_tok[rl];
        if (t >= 0)
          atomicAdd(&out[(size_t)t * DIM + col], acc[m][n][r] * s_w[rl]);
      }
    }
}

// ---------------- Aux loss (deterministic fixed-order reduction) ----------------
__global__ void k_aux(const float* __restrict__ probs, const int* __restrict__ counts,
                      float* __restrict__ aux_out) {
  __shared__ float red[256];
  __shared__ float tot[NE];
  const int tid = threadIdx.x;
  float part[NE] = {0};
  for (int n = tid; n < NTOK; n += 256) {
    #pragma unroll
    for (int e = 0; e < NE; ++e) part[e] += probs[(size_t)n * NE + e];
  }
  for (int e = 0; e < NE; ++e) {
    red[tid] = part[e];
    __syncthreads();
    for (int s = 128; s > 0; s >>= 1) {
      if (tid < s) red[tid] += red[tid + s];
      __syncthreads();
    }
    if (tid == 0) tot[e] = red[0];
    __syncthreads();
  }
  if (tid == 0) {
    float aux = 0.f;
    for (int e = 0; e < NE; ++e) {
      float f = (float)counts[e] / (float)(NTOK * TOPK);
      float p = tot[e] / (float)NTOK;
      aux += f * p;
    }
    aux_out[0] = 0.01f * (float)NE * aux;
  }
}

extern "C" void kernel_launch(void* const* d_in, const int* in_sizes, int n_in,
                              void* d_out, int out_size, void* d_ws, size_t ws_size,
                              hipStream_t stream) {
  (void)in_sizes; (void)n_in; (void)out_size; (void)ws_size;
  const float* x  = (const float*)d_in[0];
  const float* wg = (const float*)d_in[1];
  const float* w1 = (const float*)d_in[2];
  const float* w2 = (const float*)d_in[3];
  float* out = (float*)d_out;

  // Workspace carve-out (~84 MB total; dominated by H = 9216 x 4096 f16)
  char* w = (char*)d_ws;
  float* probs  = (float*)w;  w += (size_t)NTOK * NE * 4;     // 128 KB
  int*   topi   = (int*)w;    w += (size_t)NTOK * TOPK * 4;   // 32 KB
  float* topw   = (float*)w;  w += (size_t)NTOK * TOPK * 4;   // 32 KB
  int*   counts = (int*)w;    w += 256;
  int*   cursors= (int*)w;    w += 256;
  int*   offs   = (int*)w;    w += 256;
  _Float16* zbuf= (_Float16*)w; w += 256;                     // zero pad rows
  int*   tokl   = (int*)w;    w += (size_t)MAXROWS * 4;       // 36 KB
  float* tokw   = (float*)w;  w += (size_t)MAXROWS * 4;       // 36 KB
  _Float16* xh  = (_Float16*)w; w += (size_t)NTOK * DIM * 2;  // 8 MB
  _Float16* H   = (_Float16*)w;                               // 75.5 MB

  (void)hipMemsetAsync(out, 0, (size_t)NTOK * DIM * sizeof(float), stream);
  (void)hipMemsetAsync(counts, 0, 1024, stream);                 // counts+cursors+offs+zbuf
  (void)hipMemsetAsync(tokl, 0xFF, (size_t)MAXROWS * 4, stream); // padding rows = -1

  k_xcvt   <<<(NTOK * DIM) / (256 * 8), 256, 0, stream>>>(x, xh);
  k_router <<<NTOK, 128, 0, stream>>>(x, wg, probs, topi, topw, counts);
  k_scan   <<<1, 32, 0, stream>>>(counts, offs, cursors);
  k_scatter<<<NTOK / 256, 256, 0, stream>>>(topi, topw, cursors, tokl, tokw);
  k_ffn1   <<<dim3(ROWTILES, HID / BN), 256, 0, stream>>>(xh, w1, tokl, offs, zbuf, H);
  k_ffn2   <<<dim3(ROWTILES, DIM / BN), 256, 0, stream>>>(H, w2, tokl, tokw, offs, out);
  k_aux    <<<1, 256, 0, stream>>>(probs, counts, out + (size_t)NTOK * DIM);
}
